// StandardRouter_46574625358028
// MI455X (gfx1250) — compile-verified, matrix operand order checked by reading, not run on hardware
//
#include <hip/hip_runtime.h>
#include <hip/hip_bf16.h>
#include <math.h>

// ---------------------------------------------------------------------------
// MoE (top-3 of 8 experts), D=1024, H=4096, N=4096 tokens.
// route -> compact per-expert row lists -> grouped bf16 WMMA GEMMs
// (FFN1 + exact GELU, FFN2) with TDM (tensor_load_to_lds) double-buffered
// weight/activation tiles -> deterministic weighted gather-combine.
// ---------------------------------------------------------------------------

typedef __attribute__((ext_vector_type(16))) __bf16 bfx16;
typedef __attribute__((ext_vector_type(8)))  __bf16 bfx8;
typedef __attribute__((ext_vector_type(8)))  float  fx8;
typedef __attribute__((ext_vector_type(4)))  unsigned int u32x4;
typedef __attribute__((ext_vector_type(8)))  int          i32x8;
typedef __attribute__((ext_vector_type(4)))  int          i32x4;

#define D_MODEL 1024
#define N_EXP   8
#define TOPK    3
#define HIDDEN  4096

#define BM 128
#define BN 128
#define BK 64

__device__ __forceinline__ __bf16 tobf(float f) {
  unsigned u = __builtin_bit_cast(unsigned, f);
  u += 0x7FFFu + ((u >> 16) & 1u);              // round-to-nearest-even
  unsigned short s = (unsigned short)(u >> 16);
  return __builtin_bit_cast(__bf16, s);
}

__device__ __forceinline__ float gelu_exact(float x) {
  return 0.5f * x * (1.0f + erff(x * 0.70710678118654752440f));
}

// ---------------------------------------------------------------------------
// Tensor Data Mover: 2D tile (tile0 x tile1 bf16 elements) global -> LDS.
// D# layout per CDNA5 ISA:
//   g0: count=1 | lds_addr[63:32] | global_addr[120:64] | type=2[127:126]
//   g1: data_size=1(2B)[17:16] | tensor_dim0[79:48] | tensor_dim1[111:80]
//       | tile_dim0[127:112] | tile_dim1[143:128] | dim0_stride[207:160]
// Rows y >= tensor_dim1 are zero-filled by hardware (OOB read -> 0).
// ---------------------------------------------------------------------------
__device__ __forceinline__ void tdm_load_2d(unsigned lds_addr, const void* gptr,
                                            unsigned tile0, unsigned tile1,
                                            unsigned long long stride0,
                                            unsigned tdim0, unsigned tdim1) {
  unsigned long long ga = (unsigned long long)gptr;
  u32x4 g0;
  g0[0] = 1u;                                        // count=1, user mode
  g0[1] = lds_addr;                                  // LDS byte address
  g0[2] = (unsigned)(ga & 0xFFFFFFFFu);
  g0[3] = (unsigned)((ga >> 32) & 0x01FFFFFFu) | (2u << 30);  // type=2 (image)
  i32x8 g1;
  g1[0] = (int)(1u << 16);                           // data_size = 2 bytes
  g1[1] = (int)((tdim0 & 0xFFFFu) << 16);
  g1[2] = (int)((tdim0 >> 16) | ((tdim1 & 0xFFFFu) << 16));
  g1[3] = (int)((tdim1 >> 16) | (tile0 << 16));
  g1[4] = (int)(tile1 & 0xFFFFu);                    // tile_dim1 (tile_dim2=0)
  g1[5] = (int)(unsigned)(stride0 & 0xFFFFFFFFu);    // dim0 stride (elements)
  g1[6] = (int)(unsigned)((stride0 >> 32) & 0xFFFFu);
  g1[7] = 0;
  i32x4 z4 = {0, 0, 0, 0};
#if defined(__clang_major__) && (__clang_major__ >= 23)
  i32x8 z8 = {0, 0, 0, 0, 0, 0, 0, 0};
  __builtin_amdgcn_tensor_load_to_lds(g0, g1, z4, z4, z8, 0);
#else
  __builtin_amdgcn_tensor_load_to_lds(g0, g1, z4, z4, 0);
#endif
}

// Load a 16x32 bf16 A/B fragment from LDS (row stride `stride` elements).
// CDNA5 16-bit layout: lane L -> row (L&15); kBase = (L&16)?8:0;
// elements 0..7 -> k=kBase+0..7 ; 8..15 -> k=kBase+16..23 (two b128 reads).
__device__ __forceinline__ bfx16 load_frag(const __bf16* base, int lane, int stride) {
  int r = lane & 15;
  int kb = (lane & 16) ? 8 : 0;
  const __bf16* p = base + r * stride + kb;
  bfx8 lo = *(const bfx8*)(p);
  bfx8 hi = *(const bfx8*)(p + 16);
  bfx16 f;
#pragma unroll
  for (int i = 0; i < 8; ++i) { f[i] = lo[i]; f[i + 8] = hi[i]; }
  return f;
}

// ---------------------------------------------------------------------------
// 1. Gating: one wave(32) per token. logits -> softmax -> top-3 -> renorm.
// ---------------------------------------------------------------------------
__global__ __launch_bounds__(256)
void gate_kernel(const float* __restrict__ x, const float* __restrict__ gw,
                 const float* __restrict__ gb, int* __restrict__ topki,
                 float* __restrict__ topkw, float* __restrict__ probsum,
                 int* __restrict__ counts, int N) {
  int wave = threadIdx.x >> 5, lane = threadIdx.x & 31;
  int t = blockIdx.x * 8 + wave;
  if (t >= N) return;

  float part[N_EXP];
#pragma unroll
  for (int e = 0; e < N_EXP; ++e) part[e] = 0.0f;

  for (int j = 0; j < D_MODEL / 32; ++j) {
    int d = lane + j * 32;
    float xv = x[(size_t)t * D_MODEL + d];
#pragma unroll
    for (int e = 0; e < N_EXP; ++e) part[e] += xv * gw[e * D_MODEL + d];
  }
#pragma unroll
  for (int off = 16; off > 0; off >>= 1)
#pragma unroll
    for (int e = 0; e < N_EXP; ++e) part[e] += __shfl_xor(part[e], off, 32);

  if (lane == 0) {
    float p[N_EXP];
    float mx = -1e30f;
#pragma unroll
    for (int e = 0; e < N_EXP; ++e) { part[e] += gb[e]; mx = fmaxf(mx, part[e]); }
    float s = 0.0f;
#pragma unroll
    for (int e = 0; e < N_EXP; ++e) { p[e] = __expf(part[e] - mx); s += p[e]; }
    float inv = 1.0f / s;
#pragma unroll
    for (int e = 0; e < N_EXP; ++e) {
      p[e] *= inv;
      atomicAdd(&probsum[e], p[e]);
    }
    float rem[N_EXP];
#pragma unroll
    for (int e = 0; e < N_EXP; ++e) rem[e] = p[e];
    float wsel[TOPK]; int isel[TOPK];
#pragma unroll
    for (int k = 0; k < TOPK; ++k) {
      int bi = 0; float bv = rem[0];
#pragma unroll
      for (int e = 1; e < N_EXP; ++e)
        if (rem[e] > bv) { bv = rem[e]; bi = e; }   // ties -> lowest index
      wsel[k] = bv; isel[k] = bi; rem[bi] = -1.0f;
    }
    float sw = 1.0f / (wsel[0] + wsel[1] + wsel[2]);
#pragma unroll
    for (int k = 0; k < TOPK; ++k) {
      topki[t * TOPK + k] = isel[k];
      topkw[t * TOPK + k] = wsel[k] * sw;
      atomicAdd(&counts[isel[k]], 1);
    }
  }
}

// ---------------------------------------------------------------------------
// 2. Bookkeeping.
// ---------------------------------------------------------------------------
__global__ void init_small_kernel(int* counts, int* cursors, float* probsum) {
  int i = threadIdx.x;
  if (i < N_EXP) { counts[i] = 0; cursors[i] = 0; probsum[i] = 0.0f; }
}

__global__ void scan_kernel(const int* counts, int* offsets) {
  if (threadIdx.x == 0) {
    int acc = 0;
    for (int e = 0; e < N_EXP; ++e) { offsets[e] = acc; acc += counts[e]; }
  }
}

__global__ __launch_bounds__(256)
void scatter_kernel(const int* __restrict__ topki, const int* __restrict__ offsets,
                    int* __restrict__ cursors, int* __restrict__ tokenOf,
                    int* __restrict__ posOf, int total) {
  int i = blockIdx.x * blockDim.x + threadIdx.x;
  if (i >= total) return;
  int n = i / TOPK;
  int e = topki[i];
  int pos = atomicAdd(&cursors[e], 1);
  int row = offsets[e] + pos;
  tokenOf[row] = n;
  posOf[i] = row;
}

__global__ __launch_bounds__(256)
void cast_kernel(const float* __restrict__ src, __bf16* __restrict__ dst, int total4) {
  int i = blockIdx.x * blockDim.x + threadIdx.x;
  if (i >= total4) return;
  const float4 v = *(const float4*)(src + (size_t)i * 4);
  __bf16* o = dst + (size_t)i * 4;
  o[0] = tobf(v.x); o[1] = tobf(v.y); o[2] = tobf(v.z); o[3] = tobf(v.w);
}

// ---------------------------------------------------------------------------
// 3. FFN1 grouped GEMM: h = gelu(x_sel @ w1[e]^T + b1[e]), bf16 out.
//    A: gathered token rows (manual b128 copies), B: TDM double-buffered.
//    grid = (H/BN, maxRowTiles, E), block = 256 (8 waves).
// ---------------------------------------------------------------------------
__global__ __launch_bounds__(256)
void ffn1_kernel(const __bf16* __restrict__ xbf, const __bf16* __restrict__ w1bf,
                 const float* __restrict__ b1, const int* __restrict__ tokenOf,
                 const int* __restrict__ counts, const int* __restrict__ offsets,
                 __bf16* __restrict__ h) {
  const int e = blockIdx.z;
  const int cnt = counts[e];
  const int rowBase = blockIdx.y * BM;
  if (rowBase >= cnt) return;
  const int colBase = blockIdx.x * BN;
  const int base = offsets[e];

  __shared__ __align__(16) __bf16 As[2][BM * BK];
  __shared__ __align__(16) __bf16 Bs[2][BN * BK];
  __shared__ int Ts[BM];

  const int tid = threadIdx.x, lane = tid & 31, wave = tid >> 5;

  if (tid < BM) {
    int r = rowBase + tid;
    Ts[tid] = (r < cnt) ? tokenOf[base + r] : 0;  // invalid rows read token 0;
  }                                               // output store is row-guarded
  __syncthreads();

  fx8 acc[8];
#pragma unroll
  for (int nf = 0; nf < 8; ++nf)
#pragma unroll
    for (int i = 0; i < 8; ++i) acc[nf][i] = 0.0f;

  auto stageA = [&](int buf, int kk) {
#pragma unroll
    for (int i = 0; i < 4; ++i) {
      int c = tid + 256 * i;
      int row = c >> 3, cc = c & 7;
      int tok = Ts[row];
      bfx8 v = *(const bfx8*)(xbf + (size_t)tok * D_MODEL + kk + cc * 8);
      *(bfx8*)(&As[buf][row * BK + cc * 8]) = v;
    }
  };
  auto tdmB = [&](int buf, int kk) {
    tdm_load_2d((unsigned)(size_t)&Bs[buf][0],
                w1bf + ((size_t)e * HIDDEN + colBase) * D_MODEL + kk,
                BK, BN, (unsigned long long)D_MODEL, BK, BN);
  };
  auto compute = [&](int buf) {
#pragma unroll
    for (int ks = 0; ks < BK; ks += 32) {
      bfx16 a = load_frag(&As[buf][wave * 16 * BK + ks], lane, BK);
#pragma unroll
      for (int nf = 0; nf < 8; ++nf) {
        bfx16 b = load_frag(&Bs[buf][nf * 16 * BK + ks], lane, BK);
        acc[nf] = __builtin_amdgcn_wmma_f32_16x16x32_bf16(
            false, a, false, b, (short)0, acc[nf], false, false);
      }
    }
  };

  const int NIT = D_MODEL / BK;
  stageA(0, 0);
  if (wave == 0) tdmB(0, 0);
  for (int it = 0; it < NIT; ++it) {
    int cur = it & 1, nxt = cur ^ 1;
    if (wave == 0) __builtin_amdgcn_s_wait_tensorcnt(0);
    __syncthreads();                 // tile `cur` ready; prev reads finished
    if (it + 1 < NIT) {
      stageA(nxt, (it + 1) * BK);
      if (wave == 0) tdmB(nxt, (it + 1) * BK);
    }
    compute(cur);
  }

  const int n0 = lane & 15;
  const int mAdd = (lane & 16) ? 8 : 0;
#pragma unroll
  for (int nf = 0; nf < 8; ++nf) {
    int col = colBase + nf * 16 + n0;
    float bias = b1[e * HIDDEN + col];
#pragma unroll
    for (int r = 0; r < 8; ++r) {
      int row = rowBase + wave * 16 + r + mAdd;
      if (row < cnt) {
        float v = gelu_exact(acc[nf][r] + bias);
        h[(size_t)(base + row) * HIDDEN + col] = tobf(v);
      }
    }
  }
}

// ---------------------------------------------------------------------------
// 4. FFN2 grouped GEMM: out_sel = h @ w2[e]^T + b2[e], f32 out.
//    A and B tiles both via TDM, double-buffered. grid = (D/BN, rowTiles, E).
// ---------------------------------------------------------------------------
__global__ __launch_bounds__(256)
void ffn2_kernel(const __bf16* __restrict__ h, const __bf16* __restrict__ w2bf,
                 const float* __restrict__ b2, const int* __restrict__ counts,
                 const int* __restrict__ offsets, float* __restrict__ out_sel) {
  const int e = blockIdx.z;
  const int cnt = counts[e];
  const int rowBase = blockIdx.y * BM;
  if (rowBase >= cnt) return;
  const int colBase = blockIdx.x * BN;
  const int base = offsets[e];
  const unsigned validRows = (unsigned)(cnt - rowBase);   // TDM zero-fills y>=this

  __shared__ __align__(16) __bf16 As[2][BM * BK];
  __shared__ __align__(16) __bf16 Bs[2][BN * BK];

  const int tid = threadIdx.x, lane = tid & 31, wave = tid >> 5;

  fx8 acc[8];
#pragma unroll
  for (int nf = 0; nf < 8; ++nf)
#pragma unroll
    for (int i = 0; i < 8; ++i) acc[nf][i] = 0.0f;

  auto tdmA = [&](int buf, int kk) {
    tdm_load_2d((unsigned)(size_t)&As[buf][0],
                h + (size_t)(base + rowBase) * HIDDEN + kk,
                BK, BM, (unsigned long long)HIDDEN, BK, validRows);
  };
  auto tdmB = [&](int buf, int kk) {
    tdm_load_2d((unsigned)(size_t)&Bs[buf][0],
                w2bf + ((size_t)e * D_MODEL + colBase) * HIDDEN + kk,
                BK, BN, (unsigned long long)HIDDEN, BK, BN);
  };
  auto compute = [&](int buf) {
#pragma unroll
    for (int ks = 0; ks < BK; ks += 32) {
      bfx16 a = load_frag(&As[buf][wave * 16 * BK + ks], lane, BK);
#pragma unroll
      for (int nf = 0; nf < 8; ++nf) {
        bfx16 b = load_frag(&Bs[buf][nf * 16 * BK + ks], lane, BK);
        acc[nf] = __builtin_amdgcn_wmma_f32_16x16x32_bf16(
            false, a, false, b, (short)0, acc[nf], false, false);
      }
    }
  };

  const int NIT = HIDDEN / BK;
  if (wave == 0) { tdmA(0, 0); tdmB(0, 0); }
  for (int it = 0; it < NIT; ++it) {
    int cur = it & 1, nxt = cur ^ 1;
    if (wave == 0) __builtin_amdgcn_s_wait_tensorcnt(0);
    __syncthreads();
    if ((it + 1 < NIT) && wave == 0) { tdmA(nxt, (it + 1) * BK); tdmB(nxt, (it + 1) * BK); }
    compute(cur);
  }

  const int n0 = lane & 15;
  const int mAdd = (lane & 16) ? 8 : 0;
#pragma unroll
  for (int nf = 0; nf < 8; ++nf) {
    int col = colBase + nf * 16 + n0;
    float bias = b2[e * D_MODEL + col];
#pragma unroll
    for (int r = 0; r < 8; ++r) {
      int row = rowBase + wave * 16 + r + mAdd;
      if (row < cnt)
        out_sel[(size_t)(base + row) * D_MODEL + col] = acc[nf][r] + bias;
    }
  }
}

// ---------------------------------------------------------------------------
// 5. Combine (deterministic gather) + aux loss.
// ---------------------------------------------------------------------------
__global__ __launch_bounds__(256)
void combine_kernel(const float* __restrict__ out_sel, const int* __restrict__ posOf,
                    const float* __restrict__ topkw, float* __restrict__ out) {
  int n = blockIdx.x;
  int t = threadIdx.x;            // 256 threads x float4 = 1024 cols
  float ox = 0.f, oy = 0.f, oz = 0.f, ow = 0.f;
#pragma unroll
  for (int k = 0; k < TOPK; ++k) {
    int pos = posOf[n * TOPK + k];
    float w = topkw[n * TOPK + k];
    const float4 v = *(const float4*)(out_sel + (size_t)pos * D_MODEL + t * 4);
    ox += w * v.x; oy += w * v.y; oz += w * v.z; ow += w * v.w;
  }
  float4 o; o.x = ox; o.y = oy; o.z = oz; o.w = ow;
  *(float4*)(out + (size_t)n * D_MODEL + t * 4) = o;
}

__global__ void aux_kernel(const float* probsum, const int* counts,
                           float* out, int N) {
  if (threadIdx.x == 0) {
    float invN = 1.0f / (float)N;
    float s = 0.0f;
    for (int e = 0; e < N_EXP; ++e)
      s += (probsum[e] * invN) * ((float)counts[e] * invN);
    out[(size_t)N * D_MODEL] = (float)N_EXP * s;
  }
}

// ---------------------------------------------------------------------------
// Host launcher.
// ---------------------------------------------------------------------------
extern "C" void kernel_launch(void* const* d_in, const int* in_sizes, int n_in,
                              void* d_out, int out_size, void* d_ws, size_t ws_size,
                              hipStream_t stream) {
  const float* x  = (const float*)d_in[0];
  const float* gw = (const float*)d_in[1];
  const float* gb = (const float*)d_in[2];
  const float* w1 = (const float*)d_in[3];
  const float* b1 = (const float*)d_in[4];
  const float* w2 = (const float*)d_in[5];
  const float* b2 = (const float*)d_in[6];
  float* out = (float*)d_out;

  const int N = in_sizes[0] / D_MODEL;       // 4096 tokens
  const int totalRows = N * TOPK;            // 12288 selected rows
  const size_t wElems = (size_t)N_EXP * HIDDEN * D_MODEL;   // per weight tensor

  // Workspace carve-out (256B aligned).
  char* base = (char*)d_ws;
  size_t off = 0;
  auto carve = [&](size_t bytes) -> char* {
    char* p = base + off;
    off = (off + bytes + 255) & ~(size_t)255;
    return p;
  };
  __bf16* xbf     = (__bf16*)carve((size_t)N * D_MODEL * 2);
  __bf16* w1bf    = (__bf16*)carve(wElems * 2);
  __bf16* w2bf    = (__bf16*)carve(wElems * 2);
  int*    topki   = (int*)   carve((size_t)totalRows * 4);
  float*  topkw   = (float*) carve((size_t)totalRows * 4);
  int*    counts  = (int*)   carve(N_EXP * 4);
  int*    cursors = (int*)   carve(N_EXP * 4);
  int*    offsets = (int*)   carve(N_EXP * 4);
  float*  probsum = (float*) carve(N_EXP * 4);
  int*    tokenOf = (int*)   carve((size_t)totalRows * 4);
  int*    posOf   = (int*)   carve((size_t)totalRows * 4);
  __bf16* h       = (__bf16*)carve((size_t)totalRows * HIDDEN * 2);
  float*  out_sel = (float*) carve((size_t)totalRows * D_MODEL * 4);
  (void)ws_size; (void)n_in; (void)out_size;

  init_small_kernel<<<1, 32, 0, stream>>>(counts, cursors, probsum);

  gate_kernel<<<N / 8, 256, 0, stream>>>(x, gw, gb, topki, topkw, probsum, counts, N);

  scan_kernel<<<1, 32, 0, stream>>>(counts, offsets);

  scatter_kernel<<<(totalRows + 255) / 256, 256, 0, stream>>>(
      topki, offsets, cursors, tokenOf, posOf, totalRows);

  cast_kernel<<<(N * D_MODEL / 4 + 255) / 256, 256, 0, stream>>>(
      x, xbf, N * D_MODEL / 4);
  cast_kernel<<<(int)((wElems / 4 + 255) / 256), 256, 0, stream>>>(
      w1, w1bf, (int)(wElems / 4));
  cast_kernel<<<(int)((wElems / 4 + 255) / 256), 256, 0, stream>>>(
      w2, w2bf, (int)(wElems / 4));

  const int maxRowTiles = (N + BM - 1) / BM;   // worst case: one expert everywhere

  dim3 g1(HIDDEN / BN, maxRowTiles, N_EXP);
  ffn1_kernel<<<g1, 256, 0, stream>>>(xbf, w1bf, b1, tokenOf, counts, offsets, h);

  dim3 g2(D_MODEL / BN, maxRowTiles, N_EXP);
  ffn2_kernel<<<g2, 256, 0, stream>>>(h, w2bf, b2, counts, offsets, out_sel);

  combine_kernel<<<N, 256, 0, stream>>>(out_sel, posOf, topkw, out);

  aux_kernel<<<1, 32, 0, stream>>>(probsum, counts, out, N);
}
